// LocalAggregation_41154376630663
// MI455X (gfx1250) — matrix operand order repaired and checked by reading, not compiled, not run on hardware
//
#include <hip/hip_runtime.h>

// Problem constants (match reference)
#define Bq 8
#define Nq 2048
#define NSAMP 32
#define Dq 64
#define C1q 64
#define C2q 128
#define R2q (0.15f * 0.15f)
#define EPSq 1e-5f
#define CNT_INV (1.0f / (float)(Bq * Nq * NSAMP))

typedef __attribute__((ext_vector_type(16))) _Float16 v16h;
typedef __attribute__((ext_vector_type(8)))  _Float16 v8h;
typedef __attribute__((ext_vector_type(8)))  float    v8f;

__device__ __forceinline__ v8f wmma16(v16h a, v16h b, v8f c) {
    // (neg_a, A, neg_b, B, c_mod, C, reuse_a, reuse_b)
    return __builtin_amdgcn_wmma_f32_16x16x32_f16(false, a, false, b, (short)0, c, false, false);
}

__device__ __forceinline__ v16h cat8(v8h lo, v8h hi) {
    v16h r;
#pragma unroll
    for (int i = 0; i < 8; ++i) { r[i] = lo[i]; r[i + 8] = hi[i]; }
    return r;
}

// A-fragment (16-bit A 16x32 layout): lane owns row M=lane%16.
// half h = lane/16: element e -> K = 32*kt + 16h + e
__device__ __forceinline__ v16h loadAfragGlobal(const float* __restrict__ frow, int kt, int h) {
    const float* pA = frow + 32 * kt + 16 * h;       // K block {16h..16h+7}
    const float* pB = frow + 32 * kt + 16 * h + 8;   // K block {16h+8..16h+15}
    float4 x0 = *(const float4*)pA;
    float4 x1 = *(const float4*)(pA + 4);
    float4 y0 = *(const float4*)pB;
    float4 y1 = *(const float4*)(pB + 4);
    v16h r;
    r[0] = (_Float16)x0.x; r[1] = (_Float16)x0.y; r[2] = (_Float16)x0.z; r[3] = (_Float16)x0.w;
    r[4] = (_Float16)x1.x; r[5] = (_Float16)x1.y; r[6] = (_Float16)x1.z; r[7] = (_Float16)x1.w;
    r[8]  = (_Float16)y0.x; r[9]  = (_Float16)y0.y; r[10] = (_Float16)y0.z; r[11] = (_Float16)y0.w;
    r[12] = (_Float16)y1.x; r[13] = (_Float16)y1.y; r[14] = (_Float16)y1.z; r[15] = (_Float16)y1.w;
    return r;
}

// B-fragment (32x16): lane = 16*(K-half) + N; element e -> K = 16*(lane/16)+e.
// LDS weight layout: Wt[col*72 + row] (stride 72 halves -> conflict-free, 16B aligned chunks)
__device__ __forceinline__ v16h loadBfragLds(const _Float16* __restrict__ base) {
    v8h lo = *(const v8h*)base;
    v8h hi = *(const v8h*)(base + 8);
    return cat8(lo, hi);
}

// A-fragment for GEMM2 from per-wave LDS tile [16 rows][64 cols] f16
__device__ __forceinline__ v16h loadAfragLds(const _Float16* __restrict__ rowbase, int kt, int h) {
    v8h lo = *(const v8h*)(rowbase + 32 * kt + 16 * h);
    v8h hi = *(const v8h*)(rowbase + 32 * kt + 16 * h + 8);
    return cat8(lo, hi);
}

// ---------------------------------------------------------------------------
// Ball query: one wave per point. First NSAMP indices (ascending) with
// dist^2 <= R^2, padded with the first hit (self always hits).
// ---------------------------------------------------------------------------
__global__ __launch_bounds__(256) void ball_query_kernel(const float* __restrict__ pos,
                                                         int* __restrict__ idxout) {
    int w = threadIdx.x >> 5, lane = threadIdx.x & 31;
    int p = blockIdx.x * 8 + w;
    int b = p >> 11, n = p & (Nq - 1);
    const float* pb = pos + (size_t)b * Nq * 3;
    float cx = pb[n * 3 + 0], cy = pb[n * 3 + 1], cz = pb[n * 3 + 2];
    int total = 0;
    int first = n;
    for (int base = 0; base < Nq; base += 32) {
        int j = base + lane;
        float dx = pb[j * 3 + 0] - cx;
        float dy = pb[j * 3 + 1] - cy;
        float dz = pb[j * 3 + 2] - cz;
        float d2 = dx * dx + dy * dy + dz * dz;
        bool hit = d2 <= R2q;
        unsigned mask = (unsigned)__ballot(hit);
        int mypos = total + __popc(mask & ((lane == 31) ? 0x7fffffffu : ((1u << lane) - 1u)));
        if (hit && mypos < NSAMP) idxout[(size_t)p * NSAMP + mypos] = j;
        if (total == 0 && mask != 0u) first = base + (__ffs(mask) - 1);
        total += __popc(mask);
        if (total >= NSAMP) break;   // wave-uniform
    }
    if (total < NSAMP && lane >= total) idxout[(size_t)p * NSAMP + lane] = first;
}

// ---------------------------------------------------------------------------
// Fused pipeline. MODE 0: accumulate BN1 stats (sum/sumsq of h1).
//                 MODE 1: apply BN1, accumulate BN2 stats (sum/sumsq of h2).
//                 MODE 2: apply BN1+BN2, max over samples, store features.
// One wave per point; 256-thread block stages weights in LDS once.
// ---------------------------------------------------------------------------
template <int MODE>
__global__ __launch_bounds__(256) void fused_kernel(
    const float* __restrict__ position, const float* __restrict__ feature,
    const float* __restrict__ W1, const float* __restrict__ W2,
    const int* __restrict__ idxbuf,
    const float* __restrict__ a1, const float* __restrict__ b1c,
    const float* __restrict__ a2, const float* __restrict__ b2c,
    float* __restrict__ s1g, float* __restrict__ s2g,
    float* __restrict__ outfeat) {

    __shared__ float    W1x[3 * 64];          // xyz rows of W1, f32
    __shared__ _Float16 W1FT[64 * 72];        // W1 rows 3..66 transposed [col][row]
    __shared__ _Float16 W2T[128 * 72];        // W2 transposed [col][row]
    __shared__ float    xyzt[8][16 * 4];      // per-wave xyz tile
    __shared__ _Float16 h1t[8][16 * 64];      // per-wave h1' tile (GEMM1->GEMM2)
    __shared__ float    statsL[256];
    __shared__ float    maxb[8][8 * 32];

    const int tid = threadIdx.x;

    // Stage weights -> LDS (f32 -> f16 convert)
    for (int e = tid; e < 67 * 64; e += 256) {
        int r = e >> 6, c = e & 63;
        float v = W1[e];
        if (r < 3) W1x[r * 64 + c] = v;
        else       W1FT[c * 72 + (r - 3)] = (_Float16)v;
    }
    for (int e = tid; e < 64 * 128; e += 256) {
        int r = e >> 7, c = e & 127;
        W2T[c * 72 + r] = (_Float16)W2[e];
    }
    statsL[tid] = 0.0f;
    __syncthreads();

    const int w = tid >> 5, lane = tid & 31;
    const int p = blockIdx.x * 8 + w;
    const int b = p >> 11, n = p & (Nq - 1);
    const int m = lane & 15, h = lane >> 4;
    const float* posb = position + (size_t)b * Nq * 3;
    const float cx = posb[n * 3 + 0], cy = posb[n * 3 + 1], cz = posb[n * 3 + 2];

    float sAcc[4]  = {0, 0, 0, 0}, qAcc[4]  = {0, 0, 0, 0};
    float s2Acc[8] = {0, 0, 0, 0, 0, 0, 0, 0}, q2Acc[8] = {0, 0, 0, 0, 0, 0, 0, 0};
    float vmax2[8] = {0, 0, 0, 0, 0, 0, 0, 0};   // post-ReLU values are >= 0

#pragma unroll
    for (int t = 0; t < 2; ++t) {
        const int samp = t * 16 + m;
        const int j = idxbuf[(size_t)p * NSAMP + samp];
        const float* frow = feature + ((size_t)(b * Nq) + j) * Dq;

        // A fragments for GEMM1 straight from the (L2-resident) gather
        v16h aF0 = loadAfragGlobal(frow, 0, h);
        v16h aF1 = loadAfragGlobal(frow, 1, h);

        // xyz tile: lanes 0..15 own one row each
        if (h == 0) {
            float px = posb[j * 3 + 0] - cx;
            float py = posb[j * 3 + 1] - cy;
            float pz = posb[j * 3 + 2] - cz;
            float* xr = &xyzt[w][m * 4];
            xr[0] = px; xr[1] = py; xr[2] = pz;
        }
        asm volatile("s_wait_dscnt 0" ::: "memory");  // wave-internal LDS producer->consumer

        // xyz rows this lane accumulates (C layout: M = 8h + r)
        float xr0[8], xr1[8], xr2[8];
#pragma unroll
        for (int r = 0; r < 8; ++r) {
            const float* xr = &xyzt[w][(h * 8 + r) * 4];
            xr0[r] = xr[0]; xr1[r] = xr[1]; xr2[r] = xr[2];
        }

        v8f acc[4];
#pragma unroll
        for (int cb = 0; cb < 4; ++cb) {
            const int ch = cb * 16 + m;
            float w0 = W1x[0 * 64 + ch], w1 = W1x[1 * 64 + ch], w2v = W1x[2 * 64 + ch];
            v8f a = {0.f, 0.f, 0.f, 0.f, 0.f, 0.f, 0.f, 0.f};
#pragma unroll
            for (int r = 0; r < 8; ++r)
                a[r] = xr0[r] * w0 + xr1[r] * w1 + xr2[r] * w2v;   // xyz channels 0..2
            acc[cb] = a;
        }

        // GEMM1: 16x64 tile, K=64 (2 K-tiles), 4 col-blocks
#pragma unroll
        for (int cb = 0; cb < 4; ++cb) {
            const int col = cb * 16 + m;
            v16h bf0 = loadBfragLds(&W1FT[col * 72 + 0  + 16 * h]);
            v16h bf1 = loadBfragLds(&W1FT[col * 72 + 32 + 16 * h]);
            acc[cb] = wmma16(aF0, bf0, acc[cb]);
            acc[cb] = wmma16(aF1, bf1, acc[cb]);
        }

        if (MODE == 0) {
#pragma unroll
            for (int cb = 0; cb < 4; ++cb)
#pragma unroll
                for (int r = 0; r < 8; ++r) {
                    float v = acc[cb][r];
                    sAcc[cb] += v; qAcc[cb] += v * v;
                }
        } else {
            // BN1 + ReLU, write h1' tile to LDS in [row][col] f16
#pragma unroll
            for (int cb = 0; cb < 4; ++cb) {
                const int ch = cb * 16 + m;
                float aa = a1[ch], bb = b1c[ch];
#pragma unroll
                for (int r = 0; r < 8; ++r) {
                    float v = acc[cb][r] * aa + bb;
                    v = v > 0.f ? v : 0.f;
                    h1t[w][(h * 8 + r) * 64 + ch] = (_Float16)v;
                }
            }
            asm volatile("s_wait_dscnt 0" ::: "memory");

            v16h c0 = loadAfragLds(&h1t[w][m * 64], 0, h);
            v16h c1 = loadAfragLds(&h1t[w][m * 64], 1, h);

            // GEMM2: 16x128, K=64 (2 K-tiles), 8 col-blocks
#pragma unroll
            for (int cb2 = 0; cb2 < 8; ++cb2) {
                const int col = cb2 * 16 + m;
                v16h d0 = loadBfragLds(&W2T[col * 72 + 0  + 16 * h]);
                v16h d1 = loadBfragLds(&W2T[col * 72 + 32 + 16 * h]);
                v8f acc2 = {0.f, 0.f, 0.f, 0.f, 0.f, 0.f, 0.f, 0.f};
                acc2 = wmma16(c0, d0, acc2);
                acc2 = wmma16(c1, d1, acc2);
                if (MODE == 1) {
#pragma unroll
                    for (int r = 0; r < 8; ++r) {
                        float v = acc2[r];
                        s2Acc[cb2] += v; q2Acc[cb2] += v * v;
                    }
                } else {
                    float aa = a2[col], bb = b2c[col];
#pragma unroll
                    for (int r = 0; r < 8; ++r) {
                        float v = acc2[r] * aa + bb;
                        v = v > 0.f ? v : 0.f;
                        vmax2[cb2] = v > vmax2[cb2] ? v : vmax2[cb2];
                    }
                }
            }
        }
    }

    if (MODE == 0) {
#pragma unroll
        for (int cb = 0; cb < 4; ++cb) {
            const int ch = cb * 16 + m;
            atomicAdd(&statsL[ch], sAcc[cb]);        // ds_add_f32
            atomicAdd(&statsL[64 + ch], qAcc[cb]);
        }
        __syncthreads();
        if (tid < 128) atomicAdd(&s1g[tid], statsL[tid]);
    } else if (MODE == 1) {
#pragma unroll
        for (int cb2 = 0; cb2 < 8; ++cb2) {
            const int ch = cb2 * 16 + m;
            atomicAdd(&statsL[ch], s2Acc[cb2]);
            atomicAdd(&statsL[128 + ch], q2Acc[cb2]);
        }
        __syncthreads();
        atomicAdd(&s2g[tid], statsL[tid]);
    } else {
        // max over the 32 samples: in-register over 2 subtiles x 8 rows,
        // then combine lane halves through LDS.
#pragma unroll
        for (int cb2 = 0; cb2 < 8; ++cb2) maxb[w][cb2 * 32 + lane] = vmax2[cb2];
        asm volatile("s_wait_dscnt 0" ::: "memory");
        if (h == 0) {
#pragma unroll
            for (int cb2 = 0; cb2 < 8; ++cb2) {
                float v0 = maxb[w][cb2 * 32 + m];
                float v1 = maxb[w][cb2 * 32 + 16 + m];
                outfeat[(size_t)p * C2q + cb2 * 16 + m] = v0 > v1 ? v0 : v1;
            }
        }
    }
}

// sums[c], sums[nch+c] -> scale/shift: y = x*a + b
__global__ void finalize_kernel(const float* __restrict__ sums,
                                const float* __restrict__ gamma,
                                const float* __restrict__ beta,
                                float* __restrict__ aOut, float* __restrict__ bOut,
                                int nch) {
    int c = threadIdx.x;
    if (c < nch) {
        float mean = sums[c] * CNT_INV;
        float var  = sums[nch + c] * CNT_INV - mean * mean;
        float a = gamma[c] * rsqrtf(var + EPSq);
        aOut[c] = a;
        bOut[c] = beta[c] - mean * a;
    }
}

__global__ void zero_stats_kernel(float* __restrict__ s) {
    if (threadIdx.x < 384) s[threadIdx.x] = 0.0f;
}

__global__ void copy_pos_kernel(const float* __restrict__ pos, float* __restrict__ out) {
    int i = blockIdx.x * 256 + threadIdx.x;
    if (i < Bq * Nq * 3) out[i] = pos[i];
}

extern "C" void kernel_launch(void* const* d_in, const int* in_sizes, int n_in,
                              void* d_out, int out_size, void* d_ws, size_t ws_size,
                              hipStream_t stream) {
    const float* position = (const float*)d_in[0];
    const float* feature  = (const float*)d_in[1];
    const float* W1 = (const float*)d_in[2];
    const float* g1 = (const float*)d_in[3];
    const float* be1 = (const float*)d_in[4];
    const float* W2 = (const float*)d_in[5];
    const float* g2 = (const float*)d_in[6];
    const float* be2 = (const float*)d_in[7];
    float* out = (float*)d_out;

    char* ws = (char*)d_ws;
    const size_t idxBytes = (size_t)Bq * Nq * NSAMP * sizeof(int);  // 2 MB
    int*   idxbuf = (int*)ws;
    float* s1g = (float*)(ws + idxBytes);   // 128 floats (sum, sumsq) for C1
    float* s2g = s1g + 128;                 // 256 floats for C2
    float* a1  = s2g + 256;                 // 64
    float* b1c = a1 + 64;                   // 64
    float* a2  = b1c + 64;                  // 128
    float* b2c = a2 + 128;                  // 128

    float* outfeat = out + (size_t)Bq * Nq * 3;
    const int nblk = (Bq * Nq) / 8;         // 2048 blocks, 8 waves each

    zero_stats_kernel<<<1, 384, 0, stream>>>(s1g);
    ball_query_kernel<<<nblk, 256, 0, stream>>>(position, idxbuf);

    fused_kernel<0><<<nblk, 256, 0, stream>>>(position, feature, W1, W2, idxbuf,
                                              a1, b1c, a2, b2c, s1g, s2g, outfeat);
    finalize_kernel<<<1, 64, 0, stream>>>(s1g, g1, be1, a1, b1c, C1q);

    fused_kernel<1><<<nblk, 256, 0, stream>>>(position, feature, W1, W2, idxbuf,
                                              a1, b1c, a2, b2c, s1g, s2g, outfeat);
    finalize_kernel<<<1, 128, 0, stream>>>(s2g, g2, be2, a2, b2c, C2q);

    fused_kernel<2><<<nblk, 256, 0, stream>>>(position, feature, W1, W2, idxbuf,
                                              a1, b1c, a2, b2c, s1g, s2g, outfeat);

    copy_pos_kernel<<<192, 256, 0, stream>>>(position, out);
}
